// Word2Vec_66606352827121
// MI455X (gfx1250) — compile-verified, matrix operand order checked by reading, not run on hardware
//
#include <hip/hip_runtime.h>
#include <math.h>

#define VOCAB 50000
#define EMBED 512
#define NSAMP 256
#define BATCH 16384

typedef __attribute__((ext_vector_type(16))) __bf16 v16bf;
typedef __attribute__((ext_vector_type(8)))  __bf16 v8bf;
typedef __attribute__((ext_vector_type(4)))  __bf16 v4bf;
typedef __attribute__((ext_vector_type(8)))  float  v8f;

union V16 { v16bf v; v8bf h[2]; };

// ---------------------------------------------------------------------------
// Prep: gather sampled rows of nce_weight -> bf16 B-matrix in ws, precompute
// per-sampled-column correction scb[s] = bias[sid] - log(NSAMP * P(sid)),
// zero the f64 loss accumulator.
// ---------------------------------------------------------------------------
__global__ void __launch_bounds__(128)
w2v_prep(const int* __restrict__ sampled_ids,
         const float* __restrict__ nce_weight,
         const float* __restrict__ nce_bias,
         float* __restrict__ scb,
         __bf16* __restrict__ Bbf,
         double* __restrict__ accum) {
  const int s = blockIdx.x;          // 0..255
  const int t = threadIdx.x;         // 0..127, one float4 each (512 floats)
  const int sid = sampled_ids[s];

  const float4* src = (const float4*)(nce_weight + (size_t)sid * EMBED);
  v4bf* dst = (v4bf*)(Bbf + (size_t)s * EMBED);
  float4 v = src[t];
  v4bf o;
  o[0] = (__bf16)v.x; o[1] = (__bf16)v.y; o[2] = (__bf16)v.z; o[3] = (__bf16)v.w;
  dst[t] = o;

  if (t == 0) {
    float cf = (float)sid;
    float p = (logf(cf + 2.f) - logf(cf + 1.f)) / logf((float)(VOCAB + 1));
    scb[s] = nce_bias[sid] - logf((float)NSAMP * p);
  }
  if (blockIdx.x == 0 && t == 1) *accum = 0.0;   // fresh every launch
}

// ---------------------------------------------------------------------------
// Main: per block 64 batch rows (4 waves x 16 rows). Stage emb rows as bf16 in
// LDS (fused with f32 true-class dot), then bf16 WMMA GEMM vs all 256 sampled
// columns, sigmoid-CE epilogue, wave reduce, one f64 atomic per wave.
// ---------------------------------------------------------------------------
__global__ void __launch_bounds__(128)
w2v_main(const int* __restrict__ center_words,
         const int* __restrict__ target_words,
         const float* __restrict__ emb,
         const float* __restrict__ nce_weight,
         const float* __restrict__ nce_bias,
         const float* __restrict__ scb,
         const __bf16* __restrict__ Bbf,
         double* __restrict__ accum) {
  // pitch 520 bf16 = 1040 B -> 16B-aligned rows; b128 reads by 16 lanes cover
  // all 64 LDS banks exactly once (stride 4 banks per lane): conflict-free.
  __shared__ __bf16 smem[4][16][520];

  const int lane = threadIdx.x & 31;
  const int wave = threadIdx.x >> 5;
  const int r    = lane & 15;        // row within wave tile / column within n-chunk
  const int hlf  = lane >> 4;        // 0 or 1 (upper lane half)
  const int row0 = blockIdx.x * 64 + wave * 16;

  // ---- Stage A tile (f32 -> bf16 LDS) fused with true-class f32 dot ----
  const int crow = center_words[row0 + r];
  const int trow = target_words[row0 + r];
  const float4* esrc = (const float4*)(emb        + (size_t)crow * EMBED + hlf * 256);
  const float4* tsrc = (const float4*)(nce_weight + (size_t)trow * EMBED + hlf * 256);

  float dot = 0.f;
#pragma unroll 8
  for (int i = 0; i < 64; ++i) {     // 64 x float4 = 256 floats per lane
    float4 e = esrc[i];
    float4 w = tsrc[i];
    dot = fmaf(e.x, w.x, dot); dot = fmaf(e.y, w.y, dot);
    dot = fmaf(e.z, w.z, dot); dot = fmaf(e.w, w.w, dot);
    v4bf o;
    o[0] = (__bf16)e.x; o[1] = (__bf16)e.y; o[2] = (__bf16)e.z; o[3] = (__bf16)e.w;
    *(v4bf*)&smem[wave][r][hlf * 256 + i * 4] = o;
  }
  dot += __shfl_xor(dot, 16, 32);    // combine the two half-row partials

  float acc = 0.f;
  if (lane < 16) {                   // true-class loss for row r (label = 1)
    float cf = (float)trow;
    float p  = (logf(cf + 2.f) - logf(cf + 1.f)) / logf((float)(VOCAB + 1));
    float x  = dot + nce_bias[trow] - logf((float)NSAMP * p);
    acc += fmaxf(x, 0.f) - x + log1pf(expf(-fabsf(x)));
  }

  // ---- Sampled-logits GEMM: v_wmma_f32_16x16x32_bf16 ----
  // 16-bit A layout (ISA 7.12.2): lane half selects K offset 0/8 within each
  // 16-wide K group; VGPR0-3 = K[g..g+8), VGPR4-7 = K[g+16..g+24).
  const int koff = hlf * 8;
  const __bf16* arow = &smem[wave][r][0];

  for (int nch = 0; nch < 16; ++nch) {
    const __bf16* brow = Bbf + (size_t)(nch * 16 + r) * EMBED + koff;
    v8f c = {};
#pragma unroll
    for (int kb = 0; kb < EMBED; kb += 32) {
      V16 a, b;
      a.h[0] = *(const v8bf*)(arow + kb + koff);        // K [kb+koff, +8)
      a.h[1] = *(const v8bf*)(arow + kb + koff + 16);   // K [kb+16+koff, +8)
      b.h[0] = *(const v8bf*)(brow + kb);
      b.h[1] = *(const v8bf*)(brow + kb + 16);
      c = __builtin_amdgcn_wmma_f32_16x16x32_bf16(
              /*neg_a=*/false, a.v, /*neg_b=*/false, b.v,
              /*c_mod=*/(short)0, c, /*reuse_a=*/false, /*reuse_b=*/false);
    }
    // epilogue: logits for 16x16 tile; VGPR j, lane L -> (m = j + 8*(L>=16),
    // n = L&15). Sigmoid-CE with label 0.
    float sc = scb[nch * 16 + r];
#pragma unroll
    for (int j = 0; j < 8; ++j) {
      float x = c[j] + sc;
      acc += fmaxf(x, 0.f) + log1pf(expf(-fabsf(x)));
    }
  }

  // ---- wave32 reduction + global f64 accumulate ----
#pragma unroll
  for (int off = 16; off > 0; off >>= 1) acc += __shfl_xor(acc, off, 32);
  if (lane == 0) atomicAdd(accum, (double)acc);
}

__global__ void w2v_final(const double* __restrict__ accum, float* __restrict__ out) {
  out[0] = (float)(*accum * (1.0 / (double)BATCH));
}

// ---------------------------------------------------------------------------
extern "C" void kernel_launch(void* const* d_in, const int* in_sizes, int n_in,
                              void* d_out, int out_size, void* d_ws, size_t ws_size,
                              hipStream_t stream) {
  (void)in_sizes; (void)n_in; (void)out_size; (void)ws_size;
  const int*   center  = (const int*)d_in[0];
  const int*   target  = (const int*)d_in[1];
  const int*   sampled = (const int*)d_in[2];
  const float* embm    = (const float*)d_in[3];
  const float* ncew    = (const float*)d_in[4];
  const float* nceb    = (const float*)d_in[5];

  // ws layout: [0,8) f64 accum | [64,1088) f32 scb[256] | [2048, +256KB) bf16 B
  char*   ws    = (char*)d_ws;
  double* accum = (double*)ws;
  float*  scb   = (float*)(ws + 64);
  __bf16* Bbf   = (__bf16*)(ws + 2048);

  w2v_prep<<<NSAMP, 128, 0, stream>>>(sampled, ncew, nceb, scb, Bbf, accum);
  w2v_main<<<BATCH / 64, 128, 0, stream>>>(center, target, embm, ncew, nceb,
                                           scb, Bbf, accum);
  w2v_final<<<1, 1, 0, stream>>>(accum, (float*)d_out);
}
